// TEMFaithful_21973052686350
// MI455X (gfx1250) — compile-verified, matrix operand order checked by reading, not run on hardware
//
#include <hip/hip_runtime.h>
#include <hip/hip_bf16.h>

#define BB   8
#define LLEN 2048
#define DG   64
#define DX   64
#define NVOC 1024
#define NACT 4
#define LN_EPS 1e-5f

typedef __attribute__((ext_vector_type(16))) _Float16 v16h;
typedef __attribute__((ext_vector_type(8)))  _Float16 v8h;
typedef __attribute__((ext_vector_type(8)))  float    v8f;
typedef __attribute__((ext_vector_type(4)))  float    v4f;

static __device__ __forceinline__ v16h combine8(v8h lo, v8h hi) {
  v16h r;
#pragma unroll
  for (int i = 0; i < 8; ++i) { r[i] = lo[i]; r[8 + i] = hi[i]; }
  return r;
}

// Fragment loader for v_wmma_f32_16x16x32_f16 A/B operands (identical lane
// layouts): lane = {M or N} in [0,16), hi = lane>>4 selects the K-phase.
// half element e: k = 16*(e>>3) + 8*hi + (e&7)  ->  two contiguous 16B runs.
static __device__ __forceinline__ v16h load_frag_h16(const _Float16* row, int hi, int kbase) {
  v8h lo  = *(const v8h*)(row + kbase + 8 * hi);
  v8h hi8 = *(const v8h*)(row + kbase + 16 + 8 * hi);
  return combine8(lo, hi8);
}

static __device__ __forceinline__ v16h load_frag_f32cvt(const float* row, int hi, int kbase) {
  v4f a = *(const v4f*)(row + kbase + 8 * hi);
  v4f b = *(const v4f*)(row + kbase + 8 * hi + 4);
  v4f c = *(const v4f*)(row + kbase + 16 + 8 * hi);
  v4f d = *(const v4f*)(row + kbase + 16 + 8 * hi + 4);
  v16h r;
#pragma unroll
  for (int i = 0; i < 4; ++i) {
    r[i]      = (_Float16)a[i];
    r[4 + i]  = (_Float16)b[i];
    r[8 + i]  = (_Float16)c[i];
    r[12 + i] = (_Float16)d[i];
  }
  return r;
}

static __device__ __forceinline__ v8f wmma_f16(v16h a, v16h b, v8f c) {
  return __builtin_amdgcn_wmma_f32_16x16x32_f16(false, a, false, b, (short)0, c,
                                                false, false);
}

// ---------------------------------------------------------------------------
// K0: Mx = content_emb[token] * (token >= NACT), stored TRANSPOSED (B, DX, L)
// in f16 so PV B-fragments are contiguous 16B loads along K (=s).
// s is the fastest-varying thread index -> coalesced MxT stores.
// ---------------------------------------------------------------------------
__global__ void prep_mx_kernel(const int* __restrict__ tokens,
                               const float* __restrict__ emb,
                               _Float16* __restrict__ MxT) {
  int idx = blockIdx.x * blockDim.x + threadIdx.x;     // B*DX*L threads
  if (idx >= BB * DX * LLEN) return;
  int s = idx & (LLEN - 1);
  int d = (idx >> 11) & (DX - 1);
  int b = idx >> 17;
  int tok = tokens[b * LLEN + s];
  float v = (tok < NACT) ? 0.0f : emb[tok * DX + d];   // emb is L2-resident (256KB)
  MxT[(size_t)idx] = (_Float16)v;                      // idx == (b*DX + d)*LLEN + s
}

// ---------------------------------------------------------------------------
// K1: sequential action scan. g' = W_a[a] @ g on action tokens (~0.4% of
// steps), else identity. Emits Qh = beta * (g before step t)  [beta folded in
// here so the attention hot loop never touches it] and Gh = g after step t,
// f16 row-major (B,L,DG). One block per batch element, thread i owns g[i].
// ---------------------------------------------------------------------------
__global__ __launch_bounds__(64) void scan_kernel(const int* __restrict__ tokens,
                                                  const float* __restrict__ g_init,
                                                  const float* __restrict__ W_a,
                                                  const float* __restrict__ beta_p,
                                                  _Float16* __restrict__ Qh,
                                                  _Float16* __restrict__ Gh) {
  __shared__ float g[DG];
  const int b = blockIdx.x;
  const int i = threadIdx.x;
  const float beta = *beta_p;
  g[i] = g_init[i];
  __syncthreads();
  const int* tok = tokens + b * LLEN;
  for (int t = 0; t < LLEN; ++t) {
    float gi = g[i];
    Qh[((size_t)(b * LLEN + t)) * DG + i] = (_Float16)(beta * gi);
    int a = tok[t];                                    // uniform across block
    if (a < NACT) {
      const float* Wr = W_a + ((size_t)a * DG + i) * DG;
      float acc = 0.0f;
#pragma unroll 8
      for (int j = 0; j < DG; ++j) acc += Wr[j] * g[j];
      __syncthreads();
      g[i] = acc;
      __syncthreads();
      gi = acc;
    }
    Gh[((size_t)(b * LLEN + t)) * DG + i] = (_Float16)gi;
  }
}

// LayerNorm + f16 store for one 16-row tile held in C-fragment layout.
static __device__ __forceinline__ void ln_epilogue(
    const v8f o[5], int base_t, int b, int n, int hi,
    float g0, float g1, float g2, float g3,
    float b0, float b1, float b2, float b3,
    _Float16* __restrict__ Yh) {
#pragma unroll
  for (int r = 0; r < 8; ++r) {
    float l = o[4][r];                                 // row denominator (no reduce)
    float inv = (l > 0.0f) ? (1.0f / l) : 0.0f;
    float x0 = o[0][r] * inv, x1 = o[1][r] * inv;
    float x2 = o[2][r] * inv, x3 = o[3][r] * inv;
    int trow = base_t + r + 8 * hi;
    if (trow == 0) { x0 = x1 = x2 = x3 = 0.0f; }       // empty memory at t=0
    float s = x0 + x1 + x2 + x3;
#pragma unroll
    for (int off = 8; off; off >>= 1) s += __shfl_xor(s, off, 16);
    float mu = s * (1.0f / 64.0f);
    float d0 = x0 - mu, d1 = x1 - mu, d2 = x2 - mu, d3 = x3 - mu;
    float q = d0 * d0 + d1 * d1 + d2 * d2 + d3 * d3;
#pragma unroll
    for (int off = 8; off; off >>= 1) q += __shfl_xor(q, off, 16);
    float rstd = rsqrtf(q * (1.0f / 64.0f) + LN_EPS);
    _Float16* yr = Yh + ((size_t)(b * LLEN + trow)) * DX;
    yr[n]      = (_Float16)(d0 * rstd * g0 + b0);
    yr[16 + n] = (_Float16)(d1 * rstd * g1 + b1);
    yr[32 + n] = (_Float16)(d2 * rstd * g2 + b2);
    yr[48 + n] = (_Float16)(d3 * rstd * g3 + b3);
  }
}

// ---------------------------------------------------------------------------
// K2: flash attention (strict causal) + LayerNorm, LDS-free inner loop.
//  * S^T = G @ Q^T (operand swap): the S^T C-fragment IS the P A-fragment
//    lane-for-lane, so no LDS transpose anywhere.
//  * softmax denominator rides as a 5th WMMA accumulator (B = ones).
//  * one shared tile-level max stabilizer; rescale is wave-uniform and
//    skipped via a uniform branch when the max doesn't grow.
//  * TWO adjacent 16-row t-tiles per wave: every G / Mx fragment load is
//    reused by both Q tiles -> 18 wmmas per 16 b128 loads per s-block,
//    halving L2 traffic per FLOP vs one tile per wave.
// ---------------------------------------------------------------------------
__global__ __launch_bounds__(128) void attn_ln_kernel(
    const _Float16* __restrict__ Qh, const _Float16* __restrict__ Gh,
    const _Float16* __restrict__ MxT,
    const float* __restrict__ ln_g, const float* __restrict__ ln_b,
    _Float16* __restrict__ Yh) {
  const int wave = threadIdx.x >> 5;
  const int lane = threadIdx.x & 31;
  const int n    = lane & 15;
  const int hi   = lane >> 4;
  const int gw   = blockIdx.x * 4 + wave;              // 0 .. B*L/32 - 1
  const int b    = gw >> 6;                            // / (L/32)
  const int t0   = (gw & 63) << 5;                     // 32 t-rows per wave

  // Q tiles A (t0..t0+15) and B (t0+16..t0+31) as B operands of S^T.
  const _Float16* QrowA = Qh + ((size_t)(b * LLEN + t0 + n)) * DG;
  const _Float16* QrowB = QrowA + 16 * DG;
  const v16h bqA0 = load_frag_h16(QrowA, hi, 0);
  const v16h bqA1 = load_frag_h16(QrowA, hi, 32);
  const v16h bqB0 = load_frag_h16(QrowB, hi, 0);
  const v16h bqB1 = load_frag_h16(QrowB, hi, 32);

  v16h ones;
#pragma unroll
  for (int i = 0; i < 16; ++i) ones[i] = (_Float16)1.0f;

  float mtile = -3.0e38f;                              // shared running max
  v8f oA[5] = {}, oB[5] = {};                          // [0..3]=x_hat, [4]=denominator

  const int send = t0 + 31;                            // tile B needs s <= t0+30
  for (int s0 = 0; s0 < send; s0 += 32) {
    // ---- shared G fragments (A operands of S^T = G @ Q^T) ----
    const _Float16* Gr0 = Gh + ((size_t)(b * LLEN + s0 + n)) * DG;       // rows s0+n
    const _Float16* Gr1 = Gr0 + 16 * DG;                                  // rows s0+16+n
    const v16h ga00 = load_frag_h16(Gr0, hi, 0);
    const v16h ga01 = load_frag_h16(Gr0, hi, 32);
    const v16h ga10 = load_frag_h16(Gr1, hi, 0);
    const v16h ga11 = load_frag_h16(Gr1, hi, 32);
    v8f sA0 = {}, sA1 = {}, sB0 = {}, sB1 = {};
    sA0 = wmma_f16(ga00, bqA0, sA0);  sA0 = wmma_f16(ga01, bqA1, sA0);
    sA1 = wmma_f16(ga10, bqA0, sA1);  sA1 = wmma_f16(ga11, bqA1, sA1);
    sB0 = wmma_f16(ga00, bqB0, sB0);  sB0 = wmma_f16(ga01, bqB1, sB0);
    sB1 = wmma_f16(ga10, bqB0, sB1);  sB1 = wmma_f16(ga11, bqB1, sB1);
    if (s0 + 32 < send) {                              // prefetch next G block (near)
      __builtin_prefetch(Gr0 + 32 * DG, 0, 3);
      __builtin_prefetch(Gr1 + 32 * DG, 0, 3);
    }

    if (s0 + 31 >= t0) {                               // partial region (covers both tiles)
#pragma unroll
      for (int r = 0; r < 8; ++r) {
        int s = s0 + r + 8 * hi;                       // S^T rows are s
        int tA = t0 + n, tB = t0 + 16 + n;             // S^T cols are t
        if (s      >= tA) sA0[r] = -1.0e9f;
        if (s + 16 >= tA) sA1[r] = -1.0e9f;
        if (s      >= tB) sB0[r] = -1.0e9f;
        if (s + 16 >= tB) sB1[r] = -1.0e9f;
      }
    }

    // ---- shared tile-level max (wave-uniform stabilizer) ----
    float tm = fmaxf(fmaxf(sA0[0], sA1[0]), fmaxf(sB0[0], sB1[0]));
#pragma unroll
    for (int r = 1; r < 8; ++r)
      tm = fmaxf(tm, fmaxf(fmaxf(sA0[r], sA1[r]), fmaxf(sB0[r], sB1[r])));
#pragma unroll
    for (int off = 16; off; off >>= 1) tm = fmaxf(tm, __shfl_xor(tm, off, 32));
    if (tm > mtile) {                                  // uniform branch, EXEC intact
      float sc = __expf(mtile - tm);                   // 0 on first block
      mtile = tm;
#pragma unroll
      for (int f = 0; f < 5; ++f) {
#pragma unroll
        for (int r = 0; r < 8; ++r) { oA[f][r] *= sc; oB[f][r] *= sc; }
      }
    }

    // ---- P = exp(S - m): S^T C-fragment IS the P A-fragment, per lane ----
    v16h pa, pb;
#pragma unroll
    for (int r = 0; r < 8; ++r) {
      pa[r]     = (_Float16)__expf(sA0[r] - mtile);    // k = 8*hi + r
      pa[8 + r] = (_Float16)__expf(sA1[r] - mtile);    // k = 16 + 8*hi + r
      pb[r]     = (_Float16)__expf(sB0[r] - mtile);
      pb[8 + r] = (_Float16)__expf(sB1[r] - mtile);
    }

    // ---- shared Mx fragments; O += P @ Mx, denominator += P @ ones ----
    const _Float16* mxb = MxT + ((size_t)b * DX) * LLEN + s0;
    const v16h m0 = load_frag_h16(mxb + (size_t)(n)      * LLEN, hi, 0);
    const v16h m1 = load_frag_h16(mxb + (size_t)(16 + n) * LLEN, hi, 0);
    const v16h m2 = load_frag_h16(mxb + (size_t)(32 + n) * LLEN, hi, 0);
    const v16h m3 = load_frag_h16(mxb + (size_t)(48 + n) * LLEN, hi, 0);
    oA[0] = wmma_f16(pa, m0, oA[0]);  oB[0] = wmma_f16(pb, m0, oB[0]);
    oA[1] = wmma_f16(pa, m1, oA[1]);  oB[1] = wmma_f16(pb, m1, oB[1]);
    oA[2] = wmma_f16(pa, m2, oA[2]);  oB[2] = wmma_f16(pb, m2, oB[2]);
    oA[3] = wmma_f16(pa, m3, oA[3]);  oB[3] = wmma_f16(pb, m3, oB[3]);
    oA[4] = wmma_f16(pa, ones, oA[4]);
    oB[4] = wmma_f16(pb, ones, oB[4]);
  }

  const float g0 = ln_g[n], g1 = ln_g[16 + n], g2 = ln_g[32 + n], g3 = ln_g[48 + n];
  const float b0 = ln_b[n], b1 = ln_b[16 + n], b2 = ln_b[32 + n], b3 = ln_b[48 + n];
  ln_epilogue(oA, t0,      b, n, hi, g0, g1, g2, g3, b0, b1, b2, b3, Yh);
  ln_epilogue(oB, t0 + 16, b, n, hi, g0, g1, g2, g3, b0, b1, b2, b3, Yh);
}

// ---------------------------------------------------------------------------
// K3: out = Y @ W_out^T + b_out.  (B*L,64) x (64,1024) WMMA GEMM.
// One wave per 16-row x 64-vocab tile; W_out rows are L2-resident (256 KB).
// ---------------------------------------------------------------------------
__global__ __launch_bounds__(128) void proj_kernel(const _Float16* __restrict__ Yh,
                                                   const float* __restrict__ W_out,
                                                   const float* __restrict__ b_out,
                                                   float* __restrict__ out) {
  const int wave = threadIdx.x >> 5;
  const int lane = threadIdx.x & 31;
  const int n    = lane & 15;
  const int hi   = lane >> 4;
  const int gw   = blockIdx.x * 4 + wave;              // 0 .. (B*L/16)*(NVOC/64)-1
  const int rowtile = gw >> 4;                         // 16-row tile of flat (b*L+t)
  const int v0      = (gw & 15) << 6;                  // 64-wide vocab tile

  const _Float16* Yr = Yh + ((size_t)(rowtile * 16 + n)) * DX;
  const v16h a0 = load_frag_h16(Yr, hi, 0);
  const v16h a1 = load_frag_h16(Yr, hi, 32);

#pragma unroll
  for (int f = 0; f < 4; ++f) {
    int v = v0 + f * 16 + n;                           // B[k][n] = W_out[v][k]
    const float* Wr = W_out + (size_t)v * DX;
    v8f c = {};
    c = wmma_f16(a0, load_frag_f32cvt(Wr, hi, 0),  c);
    c = wmma_f16(a1, load_frag_f32cvt(Wr, hi, 32), c);
    float bias = b_out[v];
#pragma unroll
    for (int r = 0; r < 8; ++r) {
      out[((size_t)(rowtile * 16 + r + 8 * hi)) * NVOC + v] = c[r] + bias;
    }
  }
}

// ---------------------------------------------------------------------------
extern "C" void kernel_launch(void* const* d_in, const int* in_sizes, int n_in,
                              void* d_out, int out_size, void* d_ws, size_t ws_size,
                              hipStream_t stream) {
  const int*   tokens = (const int*)d_in[0];
  const float* g_init = (const float*)d_in[1];
  const float* W_a    = (const float*)d_in[2];
  const float* emb    = (const float*)d_in[3];
  const float* beta   = (const float*)d_in[4];
  const float* ln_g   = (const float*)d_in[5];
  const float* ln_b   = (const float*)d_in[6];
  const float* W_out  = (const float*)d_in[7];
  const float* b_out  = (const float*)d_in[8];
  float*       out    = (float*)d_out;

  // Workspace: 4 x (B*L*64) f16 buffers = 8 MB total (L2-resident).
  _Float16* Qh  = (_Float16*)d_ws;
  _Float16* Gh  = Qh  + (size_t)BB * LLEN * DG;
  _Float16* MxT = Gh  + (size_t)BB * LLEN * DG;
  _Float16* Yh  = MxT + (size_t)BB * LLEN * DX;

  prep_mx_kernel<<<(BB * LLEN * DX + 255) / 256, 256, 0, stream>>>(tokens, emb, MxT);
  scan_kernel<<<BB, 64, 0, stream>>>(tokens, g_init, W_a, beta, Qh, Gh);
  attn_ln_kernel<<<(BB * (LLEN / 32)) / 4, 128, 0, stream>>>(Qh, Gh, MxT,
                                                             ln_g, ln_b, Yh);
  proj_kernel<<<(BB * (LLEN / 16) * (NVOC / 64)) / 4, 128, 0, stream>>>(Yh, W_out,
                                                                        b_out, out);
}
